// Codec_60069412602067
// MI455X (gfx1250) — matrix operand order repaired and checked by reading, not compiled, not run on hardware
//
#include <hip/hip_runtime.h>
#include <hip/hip_bf16.h>

// ---------------------------------------------------------------------------
// MI455X (gfx1250) implementation, round 3.
//  - Tower 3x3 convs = implicit GEMM on v_wmma_f32_16x16x32_f16 (wave32).
//  - Activations in f16 (workload is HBM-bound: ~2GB traffic vs 51 GFLOP).
//  - LDS halo tile stored [ky][col][ci] (ci innermost) so each A half-fragment
//    is ONE 16-byte ds_load_b128; weight panel [co][k] -> two ds_load_b128.
//  - Weight panel staged global->LDS with the Tensor Data Mover
//    (tensor_load_to_lds + s_wait_tensorcnt), 1-D descriptor, issued by wave 0.
//    Dynamic shared memory is used so LDS byte offsets are compile-time
//    constants for the TDM descriptor.
// ---------------------------------------------------------------------------

typedef __attribute__((ext_vector_type(16))) _Float16 v16h;
typedef __attribute__((ext_vector_type(8)))  _Float16 v8h;
typedef __attribute__((ext_vector_type(8)))  float    v8f;

#define IMG   256
#define NBAT  16
#define PLANE_ELEMS (NBAT * 3 * IMG * IMG)        // 3,145,728 (f32 plane)
#define ACT_ELEMS   (NBAT * 12 * IMG * IMG)       // 12,582,912 (f16 act)
#define KPACK 160                                  // 144 (9 taps x 16ch) padded
#define LAYER_W_ELEMS (KPACK * 16)                 // 2560 f16 per layer

// conv kernel dynamic-LDS layout (compile-time offsets for the TDM D#)
#define TILE_HALVES (3 * 68 * 16)
#define TILE_BYTES  (TILE_HALVES * 2)              // 6528
#define WLDS_OFF    TILE_BYTES                     // 6528
#define BLDS_OFF    (WLDS_OFF + LAYER_W_ELEMS * 2) // 11648
#define SMEM_TOTAL  (BLDS_OFF + 64)                // 11712

// ---------------------------------------------------------------------------
// RCT + 2x2 polyphase split:  x (16,3,512,512) -> nw/ne/sw/se (16,3,256,256)
// ---------------------------------------------------------------------------
__global__ void prep_kernel(const float* __restrict__ x,
                            float* __restrict__ nw, float* __restrict__ ne,
                            float* __restrict__ sw, float* __restrict__ se)
{
    int idx = blockIdx.x * 256 + threadIdx.x;          // (n, i, j)
    if (idx >= NBAT * IMG * IMG) return;
    int j = idx & 255;
    int i = (idx >> 8) & 255;
    int n = idx >> 16;
    float* planes[4] = { nw, ne, sw, se };
    #pragma unroll
    for (int pi = 0; pi < 2; ++pi) {
        #pragma unroll
        for (int pj = 0; pj < 2; ++pj) {
            int sy = 2 * i + pi, sx = 2 * j + pj;
            size_t b = ((size_t)n * 3) * 512 * 512 + (size_t)sy * 512 + sx;
            float r = x[b];
            float g = x[b + 512 * 512];
            float bl = x[b + 2 * 512 * 512];
            float rr = r - g;
            float bb = bl - g;
            float gg = g + (rr + bb) * 0.25f;
            float* dst = planes[pi * 2 + pj];
            size_t o = ((size_t)(n * 3) * IMG + i) * IMG + j;
            dst[o] = rr;
            dst[o + (size_t)IMG * IMG] = gg;
            dst[o + 2 * (size_t)IMG * IMG] = bb;
        }
    }
}

// ---------------------------------------------------------------------------
// Pack tower 3x3 weights into f16 [N=16 co][K=160] panels; K = tap*16 + ci.
// ---------------------------------------------------------------------------
struct WB {
    const float* w[21];
    const float* b[21];
    int cout[21];
};

__global__ void pack_kernel(WB wb, _Float16* __restrict__ wpack,
                            float* __restrict__ bpack)
{
    int L = blockIdx.x;                 // 0..20 (tower*7 + layer)
    const float* w = wb.w[L];
    const float* b = wb.b[L];
    int cout = wb.cout[L];
    for (int i = threadIdx.x; i < LAYER_W_ELEMS; i += blockDim.x) {
        int co = i / KPACK;
        int k  = i - co * KPACK;
        float v = 0.f;
        if (k < 144 && co < cout) {
            int tap = k >> 4, ci = k & 15;
            if (ci < 12) {
                int ky = tap / 3, kx = tap - ky * 3;
                v = w[(((size_t)co * 12 + ci) * 3 + ky) * 3 + kx];
            }
        }
        wpack[(size_t)L * LAYER_W_ELEMS + i] = (_Float16)v;
    }
    if (threadIdx.x < 16)
        bpack[L * 16 + threadIdx.x] =
            (threadIdx.x < cout) ? b[threadIdx.x] : 0.f;
}

// ---------------------------------------------------------------------------
// Direct initial convs (3 input channels, small K) -> f16 tower input
// ---------------------------------------------------------------------------
__device__ inline float dconv3(const float* __restrict__ src,
                               const float* __restrict__ w,
                               int n, int y, int x, int KH, int KW,
                               int ph, int pw, int co)
{
    float s = 0.f;
    for (int ci = 0; ci < 3; ++ci)
        for (int kh = 0; kh < KH; ++kh) {
            int iy = y + kh - ph;
            if (iy < 0 || iy >= IMG) continue;
            for (int kw = 0; kw < KW; ++kw) {
                int ix = x + kw - pw;
                if (ix < 0 || ix >= IMG) continue;
                s += src[(((size_t)n * 3 + ci) * IMG + iy) * IMG + ix] *
                     w[(((size_t)co * 3 + ci) * KH + kh) * KW + kw];
            }
        }
    return s;
}

__device__ inline void init_store(_Float16* out, int n, int co, int y, int x,
                                  float s)
{
    s = s > 0.f ? s : 0.01f * s;
    out[(((size_t)n * 12 + co) * IMG + y) * IMG + x] = (_Float16)s;
}

__global__ void initA_kernel(const float* __restrict__ nw,
                             const float* __restrict__ w,
                             const float* __restrict__ bia,
                             _Float16* __restrict__ out)
{
    int idx = blockIdx.x * 256 + threadIdx.x;
    if (idx >= ACT_ELEMS) return;
    int x = idx & 255, y = (idx >> 8) & 255;
    int t = idx >> 16;
    int co = t % 12, n = t / 12;
    float s = dconv3(nw, w, n, y, x, 4, 4, 1, 1, co) + bia[co];
    init_store(out, n, co, y, x, s);
}

__global__ void initB_kernel(const float* __restrict__ nw,
                             const float* __restrict__ se,
                             const float* __restrict__ wa,
                             const float* __restrict__ wb,
                             const float* __restrict__ bia,
                             _Float16* __restrict__ out)
{
    int idx = blockIdx.x * 256 + threadIdx.x;
    if (idx >= ACT_ELEMS) return;
    int x = idx & 255, y = (idx >> 8) & 255;
    int t = idx >> 16;
    int co = t % 12, n = t / 12;
    float s = dconv3(nw, wa, n, y, x, 3, 4, 1, 1, co)
            + dconv3(se, wb, n, y, x, 4, 3, 2, 1, co) + bia[co];
    init_store(out, n, co, y, x, s);
}

__global__ void initC_kernel(const float* __restrict__ nw,
                             const float* __restrict__ se,
                             const float* __restrict__ ne,
                             const float* __restrict__ wa,
                             const float* __restrict__ wb,
                             const float* __restrict__ wc,
                             const float* __restrict__ bia,
                             _Float16* __restrict__ out)
{
    int idx = blockIdx.x * 256 + threadIdx.x;
    if (idx >= ACT_ELEMS) return;
    int x = idx & 255, y = (idx >> 8) & 255;
    int t = idx >> 16;
    int co = t % 12, n = t / 12;
    float s = dconv3(nw, wa, n, y, x, 4, 3, 1, 1, co)
            + dconv3(se, wb, n, y, x, 3, 4, 1, 2, co)
            + dconv3(ne, wc, n, y, x, 4, 4, 1, 2, co) + bia[co];
    init_store(out, n, co, y, x, s);
}

// ---------------------------------------------------------------------------
// 3x3 conv (12->12 or 12->3) as implicit GEMM on v_wmma_f32_16x16x32_f16.
// Block = 128 threads (4 waves); block tile = 64 pixels of one row x 16 ch.
// Per wave: M=16 pixels, N=16 out-channels, K=160 (5 WMMA steps of 32).
//
// For K-step s, the CDNA5 16-bit A 16x32 layout gives, per lane:
//   elements 0..7  -> tap 2s,   ci = bA..bA+7   (bA = 0 or 8 by lane group)
//   elements 8..15 -> tap 2s+1, ci = bA..bA+7
// With the tile stored [ky][col][ci] each half is one 16B ds_load_b128.
// B stored [co][k]: one lane's 16 K-values are two 16B runs, starting 32 bytes apart.
//
// Weight panel staged by the Tensor Data Mover (1-D D#, wave 0 issues,
// s_wait_tensorcnt 0 then block barrier).
//
// mode 0: bias + leaky-relu -> f16 act; mode 1: clip + (target - v) -> f32.
// ---------------------------------------------------------------------------
__global__ void __launch_bounds__(128)
conv3x3_wmma_kernel(const _Float16* __restrict__ in,     // (16,12,256,256) f16
                    const _Float16* __restrict__ resid,  // nullable, same shape
                    const _Float16* __restrict__ wpack,
                    const float* __restrict__ bpack,
                    int layer, int cout,
                    _Float16* __restrict__ outf16,
                    float* __restrict__ predOut,
                    const float* __restrict__ target,    // (16,3,256,256) f32
                    int mode)
{
    extern __shared__ char smem[];
    _Float16* tile = (_Float16*)smem;                  // [ky][col][ci], off 0
    _Float16* wlds = (_Float16*)(smem + WLDS_OFF);     // [co][k]
    float*    blds = (float*)(smem + BLDS_OFF);

    const int tid  = threadIdx.x;
    const int lane = tid & 31;
    const int wave = tid >> 5;
    const int n    = blockIdx.z;
    const int y    = blockIdx.y;
    const int x0   = blockIdx.x * 64;

    // --- stage weight panel: TDM global->LDS (fallback: 128-bit copies) ---
#if __has_builtin(__builtin_amdgcn_tensor_load_to_lds)
    if (tid < 32) {
        typedef __attribute__((ext_vector_type(4))) unsigned int u32x4;
        typedef __attribute__((ext_vector_type(4))) int i32x4;
        typedef __attribute__((ext_vector_type(8))) int i32x8;
        unsigned long long ga =
            (unsigned long long)(wpack + (size_t)layer * LAYER_W_ELEMS);
        // D# group 0: count=1, lds_addr, global_addr[56:0], type=2 (image)
        u32x4 g0;
        g0[0] = 1u;
        g0[1] = (unsigned)WLDS_OFF;
        g0[2] = (unsigned)(ga & 0xffffffffull);
        g0[3] = (unsigned)((ga >> 32) & 0x1ffffffull) | (2u << 30);
        // D# group 1: mask=0, data_size=1 (2B), 1-D tile of 2560 elements
        i32x8 g1;
        g1[0] = 1 << 16;                                  // data_size = 2B
        g1[1] = (int)((LAYER_W_ELEMS & 0xffff) << 16);    // tensor_dim0 lo
        g1[2] = (int)((LAYER_W_ELEMS >> 16) & 0xffff)     // tensor_dim0 hi
              | (1 << 16);                                // tensor_dim1 = 1
        g1[3] = (int)((LAYER_W_ELEMS & 0xffff) << 16);    // tile_dim0
        g1[4] = 0;                                        // tile_dim1/2 unused
        g1[5] = (int)LAYER_W_ELEMS;                       // tensor_dim0_stride
        g1[6] = 0;
        g1[7] = 0;
        i32x4 z4 = (i32x4)0;
#if __clang_major__ >= 23
        i32x8 z8 = (i32x8)0;
        __builtin_amdgcn_tensor_load_to_lds(g0, g1, z4, z4, z8, 0);
#else
        __builtin_amdgcn_tensor_load_to_lds(g0, g1, z4, z4, 0);
#endif
        __builtin_amdgcn_s_wait_tensorcnt(0);
    }
#else
    {
        const uint4* src = (const uint4*)(wpack + (size_t)layer * LAYER_W_ELEMS);
        uint4* dst = (uint4*)wlds;
        for (int i = tid; i < LAYER_W_ELEMS / 8; i += 128)
            dst[i] = src[i];
    }
#endif
    if (tid < 16) blds[tid] = bpack[layer * 16 + tid];

    // --- stage input halo tile: rows y-1..y+1, cols x0-1..x0+64, 16 ch ---
    // col fastest for coalesced global reads.
    for (int i = tid; i < 3 * 16 * 66; i += 128) {
        int col = i % 66;
        int t2  = i / 66;
        int ci  = t2 & 15;
        int row = t2 >> 4;
        int iy = y + row - 1;
        int ix = x0 - 1 + col;
        _Float16 v = (_Float16)0.f;
        if (ci < 12 && iy >= 0 && iy < IMG && ix >= 0 && ix < IMG) {
            size_t gi = (((size_t)n * 12 + ci) * IMG + iy) * IMG + ix;
            v = in[gi];
            if (resid) v = (_Float16)((float)v + (float)resid[gi]);
        }
        tile[(row * 68 + col) * 16 + ci] = v;
    }
    __syncthreads();

    // CDNA5 prefetch of next output row's tile (global_prefetch_b8)
    if (y + 2 < IMG)
        __builtin_prefetch(&in[(((size_t)n * 12) * IMG + (y + 2)) * IMG + x0],
                           0, 1);

    const int m    = lane & 15;       // A-matrix row / D column index
    const int hi   = lane >> 4;       // 0 or 1
    const int bA   = hi * 8;          // A ci-base per 16-bit A 16x32 layout
    const int kbB  = hi * 8;          // B K-base
    const int colB = m;               // out channel for B/D
    const int xloc = wave * 16 + m;   // pixel position within block tile

    v8f acc = {};
    #pragma unroll
    for (int s = 0; s < 5; ++s) {
        // ---- A fragment: two 16B LDS loads (fixed tap, contiguous ci) ----
        v8h alo, ahi = {};
        {
            int tap = 2 * s;                       // always < 9
            int ky = tap / 3, kx = tap - ky * 3;
            alo = *(const v8h*)&tile[(ky * 68 + (xloc + kx)) * 16 + bA];
        }
        if (2 * s + 1 < 9) {
            int tap = 2 * s + 1;
            int ky = tap / 3, kx = tap - ky * 3;
            ahi = *(const v8h*)&tile[(ky * 68 + (xloc + kx)) * 16 + bA];
        }
        v16h a = __builtin_shufflevector(alo, ahi, 0, 1, 2, 3, 4, 5, 6, 7,
                                         8, 9, 10, 11, 12, 13, 14, 15);

        // ---- B fragment: two 16B LDS loads (k = kbB..+7 and 16+kbB..+7 at fixed co) ----
        const int wbase = colB * KPACK + 32 * s + kbB;
        v8h blo = *(const v8h*)&wlds[wbase];
        v8h bhi = *(const v8h*)&wlds[wbase + 16];
        v16h b = __builtin_shufflevector(blo, bhi, 0, 1, 2, 3, 4, 5, 6, 7,
                                         8, 9, 10, 11, 12, 13, 14, 15);

        acc = __builtin_amdgcn_wmma_f32_16x16x32_f16(
            false, a, false, b, (short)0, acc, false, false);
    }

    // --- epilogue: lane holds channel colB, rows M = r + hi*8 ---
    const int co    = colB;
    const int xbase = x0 + wave * 16 + hi * 8;
    if (mode == 0) {
        if (co < 12) {
            float bia = blds[co];
            #pragma unroll
            for (int r = 0; r < 8; ++r) {
                float v = acc[r] + bia;
                v = v > 0.f ? v : 0.01f * v;
                outf16[(((size_t)n * 12 + co) * IMG + y) * IMG + (xbase + r)] =
                    (_Float16)v;
            }
        }
    } else {
        if (co < 3) {
            float bia = blds[co];
            #pragma unroll
            for (int r = 0; r < 8; ++r) {
                float v = acc[r] + bia;
                v = fminf(1.f, fmaxf(-1.f, v));
                size_t ti = (((size_t)n * 3 + co) * IMG + y) * IMG + (xbase + r);
                predOut[ti] = target[ti] - v;
            }
        }
    }
}

// ---------------------------------------------------------------------------
// Reductions
// ---------------------------------------------------------------------------
__global__ void zero_kernel(float* __restrict__ accum,
                            unsigned* __restrict__ hist, int histN)
{
    int i = blockIdx.x * 256 + threadIdx.x;
    if (i < 2) accum[i] = 0.f;
    if (i < histN) hist[i] = 0u;
}

__global__ void sumsq_kernel(const float* __restrict__ a, int nElem,
                             float* __restrict__ accum)
{
    __shared__ float red[256];
    float s = 0.f;
    for (int i = blockIdx.x * 256 + threadIdx.x; i < nElem;
         i += gridDim.x * 256) {
        float v = a[i];
        s += v * v;
    }
    red[threadIdx.x] = s;
    __syncthreads();
    for (int o = 128; o > 0; o >>= 1) {
        if (threadIdx.x < o) red[threadIdx.x] += red[threadIdx.x + o];
        __syncthreads();
    }
    if (threadIdx.x == 0) atomicAdd(accum, red[0]);
}

__global__ void hist_kernel(const float* __restrict__ pred,
                            unsigned* __restrict__ hist)   // 48*256 bins
{
    __shared__ unsigned h[256];
    h[threadIdx.x] = 0u;
    __syncthreads();
    int ch = blockIdx.y;                   // n*3 + c, 0..47
    const float* p = pred + (size_t)ch * IMG * IMG;
    for (int i = blockIdx.x * 256 + threadIdx.x; i < IMG * IMG;
         i += gridDim.x * 256) {
        float xf = fmodf(p[i] + 1.f, 2.f) - 1.f;
        bool inr = (xf >= -1.f) && (xf <= 1.f);
        int bin = (int)floorf((xf + 1.f) * 128.f);
        bin = bin < 0 ? 0 : (bin > 255 ? 255 : bin);
        if (inr) atomicAdd(&h[bin], 1u);
    }
    __syncthreads();
    if (h[threadIdx.x]) atomicAdd(&hist[ch * 256 + threadIdx.x],
                                  h[threadIdx.x]);
}

__global__ void entropy_kernel(const unsigned* __restrict__ hist,
                               float* __restrict__ out_scalar)
{
    __shared__ float red[256];
    float e = 0.f;
    for (int i = threadIdx.x; i < 48 * 256; i += 256) {
        float prob = (float)hist[i] / 65536.f;
        if (prob > 0.f) e += -prob * log2f(prob);
    }
    red[threadIdx.x] = e;
    __syncthreads();
    for (int o = 128; o > 0; o >>= 1) {
        if (threadIdx.x < o) red[threadIdx.x] += red[threadIdx.x + o];
        __syncthreads();
    }
    if (threadIdx.x == 0) out_scalar[0] = red[0] / (8.f * 48.f);
}

__global__ void finalize_kernel(const float* __restrict__ accum,
                                float* __restrict__ out)
{
    if (threadIdx.x == 0) {
        out[0] = 255.f * sqrtf(accum[0] / 9437184.f);
        out[1] = 255.f * sqrtf(accum[1] / 9437184.f);
    }
}

// ---------------------------------------------------------------------------
// Host orchestration
// ---------------------------------------------------------------------------
extern "C" void kernel_launch(void* const* d_in, const int* in_sizes, int n_in,
                              void* d_out, int out_size, void* d_ws,
                              size_t ws_size, hipStream_t stream)
{
    (void)in_sizes; (void)n_in; (void)out_size; (void)ws_size;
    const float* x     = (const float*)d_in[0];
    const float* A0a_w = (const float*)d_in[1];
    const float* A0a_b = (const float*)d_in[2];
    const float* B0a_w = (const float*)d_in[3];
    const float* B0b_w = (const float*)d_in[4];
    const float* B0b_b = (const float*)d_in[5];
    const float* C0a_w = (const float*)d_in[6];
    const float* C0b_w = (const float*)d_in[7];
    const float* C0c_w = (const float*)d_in[8];
    const float* C0c_b = (const float*)d_in[9];
    // A_ws 10..16, A_bs 17..23, B_ws 24..30, B_bs 31..37, C_ws 38..44, C_bs 45..51

    float* out = (float*)d_out;

    // --- workspace carve ---
    char* base = (char*)d_ws;
    size_t off = 0;
    auto take = [&](size_t bytes) -> char* {
        char* p = base + off;
        off = (off + bytes + 255) & ~(size_t)255;
        return p;
    };
    float* nw = (float*)take(PLANE_ELEMS * 4);
    float* ne = (float*)take(PLANE_ELEMS * 4);
    float* sw = (float*)take(PLANE_ELEMS * 4);
    float* se = (float*)take(PLANE_ELEMS * 4);
    _Float16* tbuf = (_Float16*)take(ACT_ELEMS * 2);
    _Float16* h0   = (_Float16*)take(ACT_ELEMS * 2);
    _Float16* h1   = (_Float16*)take(ACT_ELEMS * 2);
    float* pred    = (float*)take(PLANE_ELEMS * 4);
    _Float16* wpack = (_Float16*)take(21 * LAYER_W_ELEMS * 2);
    float* bpack    = (float*)take(21 * 16 * 4);
    float* accum    = (float*)take(256);
    unsigned* hist  = (unsigned*)take(3 * 48 * 256 * 4);

    // --- zero accumulators + histograms ---
    int histN = 3 * 48 * 256;
    zero_kernel<<<(histN + 255) / 256, 256, 0, stream>>>(accum, hist, histN);

    // --- RCT + split ---
    prep_kernel<<<(NBAT * IMG * IMG + 255) / 256, 256, 0, stream>>>(
        x, nw, ne, sw, se);

    // --- pack tower weights ---
    WB wb;
    for (int T = 0; T < 3; ++T) {
        int wsBase = 10 + T * 14;      // T_ws start
        int bsBase = 17 + T * 14;      // T_bs start
        for (int L = 0; L < 7; ++L) {
            wb.w[T * 7 + L] = (const float*)d_in[wsBase + L];
            wb.b[T * 7 + L] = (const float*)d_in[bsBase + L];
            wb.cout[T * 7 + L] = (L == 6) ? 3 : 12;
        }
    }
    pack_kernel<<<21, 256, 0, stream>>>(wb, wpack, bpack);

    // --- loss0: RMSE over (se, ne, sw) ---
    sumsq_kernel<<<2048, 256, 0, stream>>>(se, PLANE_ELEMS, accum + 1);
    sumsq_kernel<<<2048, 256, 0, stream>>>(ne, PLANE_ELEMS, accum + 1);
    sumsq_kernel<<<2048, 256, 0, stream>>>(sw, PLANE_ELEMS, accum + 1);

    dim3 gconv(4, IMG, NBAT);
    int initBlocks = (ACT_ELEMS + 255) / 256;

    for (int T = 0; T < 3; ++T) {
        if (T == 0)
            initA_kernel<<<initBlocks, 256, 0, stream>>>(nw, A0a_w, A0a_b,
                                                         tbuf);
        else if (T == 1)
            initB_kernel<<<initBlocks, 256, 0, stream>>>(nw, se, B0a_w, B0b_w,
                                                         B0b_b, tbuf);
        else
            initC_kernel<<<initBlocks, 256, 0, stream>>>(nw, se, ne, C0a_w,
                                                         C0b_w, C0c_w, C0c_b,
                                                         tbuf);

        const _Float16* src = tbuf;
        _Float16* dst = h0;
        for (int L = 0; L < 6; ++L) {
            const _Float16* rsd = (L == 5) ? tbuf : nullptr;
            conv3x3_wmma_kernel<<<gconv, 128, SMEM_TOTAL, stream>>>(
                src, rsd, wpack, bpack, T * 7 + L, 12, dst, nullptr, nullptr,
                0);
            src = dst;
            dst = (dst == h0) ? h1 : h0;
        }
        const float* tgt = (T == 0) ? se : (T == 1) ? ne : sw;
        conv3x3_wmma_kernel<<<gconv, 128, SMEM_TOTAL, stream>>>(
            src, nullptr, wpack, bpack, T * 7 + 6, 3, nullptr, pred, tgt, 1);

        sumsq_kernel<<<2048, 256, 0, stream>>>(pred, PLANE_ELEMS, accum + 0);
        hist_kernel<<<dim3(16, 48), 256, 0, stream>>>(pred,
                                                      hist + T * 48 * 256);
        entropy_kernel<<<1, 256, 0, stream>>>(hist + T * 48 * 256,
                                              out + 2 + T);
    }

    finalize_kernel<<<1, 32, 0, stream>>>(accum, out);
}
